// MinGRULayer_19207093747863
// MI455X (gfx1250) — matrix operand order repaired and checked by reading, not compiled, not run on hardware
//
#include <hip/hip_runtime.h>

#define D_MODEL 1024
#define BATCH   4
#define SEQ     8192
#define MTOT    (BATCH * SEQ)      // 32768

#define BM 128
#define BN 64
#define BK 32
#define PADK 40                    // row stride 80B: 16B-aligned for B128 DMA, breaks bank conflicts
#define NKSTEP (D_MODEL / BK)      // 32

#define CHUNK  128
#define NCHUNK (SEQ / CHUNK)       // 64
#define NPHA   (BATCH * NCHUNK * D_MODEL)  // 262144
#define NCHAN  (BATCH * D_MODEL)           // 4096

typedef __attribute__((ext_vector_type(16))) __bf16 v16bf;
typedef __attribute__((ext_vector_type(8)))  float  v8f;

union BFrag {
    v16bf v;
    unsigned short u[16];
};

__device__ __forceinline__ unsigned short f2bf(float f) {
    unsigned int u = __float_as_uint(f);
    unsigned int r = 0x7FFFu + ((u >> 16) & 1u);   // round-to-nearest-even
    return (unsigned short)((u + r) >> 16);
}

// async global->LDS DMA (gfx1250): 16 bytes per lane, tracked by ASYNCcnt
__device__ __forceinline__ void async_b128(unsigned int lds_off,
                                           unsigned long long gaddr) {
    asm volatile("global_load_async_to_lds_b128 %0, %1, off"
                 :: "v"(lds_off), "v"(gaddr) : "memory");
}
__device__ __forceinline__ void wait_async0() {
    asm volatile("s_wait_asynccnt 0x0" ::: "memory");
}

// ---------------- Kernel 1: x f32 -> bf16 (elementwise) ----------------
__global__ void cvt_x(const float* __restrict__ X,
                      unsigned short* __restrict__ Xb, int n) {
    int i = blockIdx.x * blockDim.x + threadIdx.x;
    if (i < n) Xb[i] = f2bf(X[i]);
}

// ---------------- Kernel 2: weights f32 -> bf16, transposed to [n][k] ----------------
__global__ void cvt_wt(const float* __restrict__ W,
                       unsigned short* __restrict__ Wt, int n) {
    int i = blockIdx.x * blockDim.x + threadIdx.x;
    if (i < n) {
        int nn = i >> 10;          // output column (row of Wt)
        int kk = i & (D_MODEL - 1);
        Wt[i] = f2bf(W[(size_t)kk * D_MODEL + nn]);
    }
}

// ---------------- Kernel 3: fused dual GEMM + gating epilogue ----------------
// z_pre = X*Wz + bz ; h_til = X*Wh + bh ; z = sigmoid(z_pre)
// Aout = 1 - z ; Bout = z * h_til
// Tiles DMA'd to LDS via global_load_async_to_lds_b128, double buffered.
#define SM_A(b)  ((unsigned)(b) * BM * PADK)                      // ushort index
#define SM_BZ(b) (2 * BM * PADK + (unsigned)(b) * BN * PADK)
#define SM_BH(b) (2 * BM * PADK + 2 * BN * PADK + (unsigned)(b) * BN * PADK)
#define SMEM_USHORTS (2 * BM * PADK + 4 * BN * PADK)              // 20480 ushorts = 40960 B

__global__ __launch_bounds__(256)
void mingru_gemm(const unsigned short* __restrict__ Xb,
                 const unsigned short* __restrict__ Wzt,
                 const unsigned short* __restrict__ Wht,
                 const float* __restrict__ bz,
                 const float* __restrict__ bh,
                 float* __restrict__ Aout,
                 float* __restrict__ Bout) {
    extern __shared__ unsigned short smem[];

    const int m0   = blockIdx.y * BM;
    const int n0   = blockIdx.x * BN;
    const int t    = threadIdx.x;
    const int lane = t & 31;
    const int wave = t >> 5;
    const int wrow = wave >> 1;   // 0..3 -> 32-row strip
    const int wcol = wave & 1;    // 0..1 -> 32-col strip
    const int half = lane >> 4;
    const int l15  = lane & 15;

    const unsigned long long xAddr  = (unsigned long long)Xb;
    const unsigned long long wzAddr = (unsigned long long)Wzt;
    const unsigned long long whAddr = (unsigned long long)Wht;

    // per-thread DMA assignments (bytes)
    const int rA  = t >> 1;           // A row 0..127
    const int sA  = (t & 1) * 32;     // 32B segment within 64B row slice
    const int rB  = t >> 2;           // B row 0..63
    const int sB  = (t & 3) * 16;     // 16B segment within 64B row

    v8f accz[2][2] = {};
    v8f acch[2][2] = {};

    // ---- prologue: DMA tile 0 into buffer 0 ----
    {
        unsigned long long ga = xAddr + ((unsigned long long)(m0 + rA) * D_MODEL) * 2ull + sA;
        async_b128(SM_A(0) * 2 + rA * PADK * 2 + sA,      ga);
        async_b128(SM_A(0) * 2 + rA * PADK * 2 + sA + 16, ga + 16);
        unsigned long long gb = ((unsigned long long)(n0 + rB) * D_MODEL) * 2ull + sB;
        async_b128(SM_BZ(0) * 2 + rB * PADK * 2 + sB, wzAddr + gb);
        async_b128(SM_BH(0) * 2 + rB * PADK * 2 + sB, whAddr + gb);
    }

    for (int kt = 0; kt < NKSTEP; ++kt) {
        const int cur = kt & 1;
        wait_async0();        // our wave's DMAs (into buffer `cur`) have landed
        __syncthreads();      // all waves' DMAs landed; buffer cur^1 free to overwrite

        // ---- issue DMA for next K tile into the other buffer ----
        if (kt + 1 < NKSTEP) {
            const int nxt = cur ^ 1;
            const unsigned long long kb = (unsigned long long)(kt + 1) * BK * 2ull;
            unsigned long long ga = xAddr + ((unsigned long long)(m0 + rA) * D_MODEL) * 2ull + kb + sA;
            async_b128(SM_A(nxt) * 2 + rA * PADK * 2 + sA,      ga);
            async_b128(SM_A(nxt) * 2 + rA * PADK * 2 + sA + 16, ga + 16);
            unsigned long long gb = ((unsigned long long)(n0 + rB) * D_MODEL) * 2ull + kb + sB;
            async_b128(SM_BZ(nxt) * 2 + rB * PADK * 2 + sB, wzAddr + gb);
            async_b128(SM_BH(nxt) * 2 + rB * PADK * 2 + sB, whAddr + gb);
        }

        // ---- fragments from buffer cur, then 8 back-to-back WMMAs ----
        const unsigned short* Al = smem + SM_A(cur);
        const unsigned short* Bz = smem + SM_BZ(cur);
        const unsigned short* Bh = smem + SM_BH(cur);

        BFrag af[2], bzf[2], bhf[2];
        #pragma unroll
        for (int mi = 0; mi < 2; ++mi) {
            int m = wrow * 32 + mi * 16 + l15;
            #pragma unroll
            for (int j = 0; j < 16; ++j) {
                int v = j >> 1, p = j & 1;
                // 16-bit A layout: VGPR0-3 -> K = half*8 + 2v+p ; VGPR4-7 -> +16
                int k = (v < 4) ? (half * 8 + 2 * v + p)
                                : (16 + half * 8 + 2 * (v - 4) + p);
                af[mi].u[j] = Al[m * PADK + k];
            }
        }
        #pragma unroll
        for (int ni = 0; ni < 2; ++ni) {
            int n = wcol * 32 + ni * 16 + l15;
            #pragma unroll
            for (int j = 0; j < 16; ++j) {
                int v = j >> 1, p = j & 1;
                // 16-bit B layout: VGPR0-3 -> K = half*8 + 2v+p ; VGPR4-7 -> +16
                int k = (v < 4) ? (half * 8 + 2 * v + p)
                                : (16 + half * 8 + 2 * (v - 4) + p);
                bzf[ni].u[j] = Bz[n * PADK + k];
                bhf[ni].u[j] = Bh[n * PADK + k];
            }
        }
        #pragma unroll
        for (int mi = 0; mi < 2; ++mi) {
            #pragma unroll
            for (int ni = 0; ni < 2; ++ni) {
                accz[mi][ni] = __builtin_amdgcn_wmma_f32_16x16x32_bf16(
                    false, af[mi].v, false, bzf[ni].v, (short)0,
                    accz[mi][ni], false, false);
                acch[mi][ni] = __builtin_amdgcn_wmma_f32_16x16x32_bf16(
                    false, af[mi].v, false, bhf[ni].v, (short)0,
                    acch[mi][ni], false, false);
            }
        }
    }

    // Epilogue: bias + sigmoid gating, write a -> ws, b -> d_out
    #pragma unroll
    for (int mi = 0; mi < 2; ++mi) {
        #pragma unroll
        for (int ni = 0; ni < 2; ++ni) {
            int gn = n0 + wcol * 32 + ni * 16 + l15;
            float bzv = bz[gn];
            float bhv = bh[gn];
            #pragma unroll
            for (int v = 0; v < 8; ++v) {
                int gm = m0 + wrow * 32 + mi * 16 + v + half * 8;
                float zp = accz[mi][ni][v] + bzv;
                float ht = acch[mi][ni][v] + bhv;
                float z  = 1.0f / (1.0f + __expf(-zp));
                size_t idx = (size_t)gm * D_MODEL + gn;
                Aout[idx] = 1.0f - z;
                Bout[idx] = z * ht;
            }
        }
    }
}

// ---------------- Kernel 4: scan phase A (per-chunk composition) ----------------
__global__ void scan_phaseA(const float* __restrict__ Aarr,
                            const float* __restrict__ Barr,
                            float* __restrict__ cA, float* __restrict__ cB) {
    int tid = blockIdx.x * blockDim.x + threadIdx.x;   // 262144 threads
    int d   = tid & (D_MODEL - 1);
    int c   = (tid >> 10) & (NCHUNK - 1);
    int bat = tid >> 16;
    size_t base = ((size_t)bat * SEQ + (size_t)c * CHUNK) * D_MODEL + d;
    float pa = 1.0f, pb = 0.0f;
    for (int i = 0; i < CHUNK; ++i) {
        float a = Aarr[base];
        float b = Barr[base];
        pa *= a;
        pb = a * pb + b;
        base += D_MODEL;
    }
    cA[tid] = pa;
    cB[tid] = pb;
}

// ---------------- Kernel 5: scan phase B (inter-chunk serial scan) ----------------
__global__ void scan_phaseB(const float* __restrict__ cA,
                            const float* __restrict__ cB,
                            float* __restrict__ cH) {
    int tid = blockIdx.x * blockDim.x + threadIdx.x;   // 4096 threads
    int d   = tid & (D_MODEL - 1);
    int bat = tid >> 10;
    float h = 0.0f;
    for (int c = 0; c < NCHUNK; ++c) {
        size_t i = ((size_t)bat * NCHUNK + c) * D_MODEL + d;
        cH[i] = h;                 // exclusive prefix: carry into chunk c
        h = cA[i] * h + cB[i];
    }
}

// ---------------- Kernel 6: scan phase C (apply carries, write h in place) ----------------
__global__ void scan_phaseC(const float* __restrict__ Aarr,
                            float* __restrict__ BH,     // holds b, overwritten with h
                            const float* __restrict__ cH) {
    int tid = blockIdx.x * blockDim.x + threadIdx.x;   // 262144 threads
    int d   = tid & (D_MODEL - 1);
    int c   = (tid >> 10) & (NCHUNK - 1);
    int bat = tid >> 16;
    size_t base = ((size_t)bat * SEQ + (size_t)c * CHUNK) * D_MODEL + d;
    float h = cH[tid];
    for (int i = 0; i < CHUNK; ++i) {
        float a = Aarr[base];
        float b = BH[base];
        h = a * h + b;
        BH[base] = h;
        base += D_MODEL;
    }
}

extern "C" void kernel_launch(void* const* d_in, const int* in_sizes, int n_in,
                              void* d_out, int out_size, void* d_ws, size_t ws_size,
                              hipStream_t stream) {
    const float* x  = (const float*)d_in[0];
    const float* Wz = (const float*)d_in[1];
    const float* bz = (const float*)d_in[2];
    const float* Wh = (const float*)d_in[3];
    const float* bh = (const float*)d_in[4];
    float* out = (float*)d_out;

    const size_t MiB = 1024ull * 1024ull;
    char* ws = (char*)d_ws;
    const size_t W_ELEMS = (size_t)D_MODEL * D_MODEL;          // 1M
    const size_t X_ELEMS = (size_t)MTOT * D_MODEL;             // 32M

    unsigned short* Wzt = (unsigned short*)ws;                 // 2 MiB
    unsigned short* Wht = (unsigned short*)(ws + 2 * MiB);     // 2 MiB
    unsigned short* Xb  = (unsigned short*)(ws + 4 * MiB);     // 64 MiB
    float* Aarr = (float*)(ws + 68 * MiB);                     // 128 MiB (a = 1-z)
    char*  ws2  = ws + 196 * MiB;
    float* cA = (float*)ws2;                                   // 1 MiB
    float* cB = (float*)(ws2 + (size_t)NPHA * 4);
    float* cH = (float*)(ws2 + 2 * (size_t)NPHA * 4);

    // 1) format prep: weights -> bf16 transposed, x -> bf16
    cvt_wt<<<(int)(W_ELEMS / 256), 256, 0, stream>>>(Wz, Wzt, (int)W_ELEMS);
    cvt_wt<<<(int)(W_ELEMS / 256), 256, 0, stream>>>(Wh, Wht, (int)W_ELEMS);
    cvt_x<<<(int)(X_ELEMS / 256), 256, 0, stream>>>(x, Xb, (int)X_ELEMS);

    // 2) fused dual GEMM + gating (b goes straight into d_out)
    dim3 grid(D_MODEL / BN, MTOT / BM);   // 16 x 256
    mingru_gemm<<<grid, 256, SMEM_USHORTS * 2, stream>>>(Xb, Wzt, Wht, bz, bh, Aarr, out);

    // 3) chunked parallel scan
    scan_phaseA<<<NPHA / 256, 256, 0, stream>>>(Aarr, out, cA, cB);
    scan_phaseB<<<NCHAN / 256, 256, 0, stream>>>(cA, cB, cH);
    scan_phaseC<<<NPHA / 256, 256, 0, stream>>>(Aarr, out, cH);
}
